// TransformerAggregator_85220741087352
// MI455X (gfx1250) — matrix operand order repaired and checked by reading, not compiled, not run on hardware
//
#include <hip/hip_runtime.h>

// ---------------------------------------------------------------------------
// TransformerAggregator for MI455X (gfx1250, wave32, WMMA bf16)
// B=32, L=1000, S=1001 (CLS + points), Sp=1024 padded, D=512, H=8, hd=64,
// DFF=1024, 2 layers. GEMMs on v_wmma_f32_16x16x32_bf16; tiles staged with
// global_load_async_to_lds_b128 (ASYNCcnt) in a double-buffered pipeline.
// ---------------------------------------------------------------------------

#define BATCH 32
#define LPTS  1000
#define SEQ   1001
#define SP    1024
#define DM    512
#define NH    8
#define HD    64
#define DFF2  1024
#define MROWS (BATCH * SP)   // 32768

typedef __attribute__((ext_vector_type(16))) __bf16 v16bf;
typedef __attribute__((ext_vector_type(8)))  float  v8f;

union FragBF { unsigned int u[8]; v16bf h; };
union AccF   { v8f v; float f[8]; };

__device__ __forceinline__ unsigned short f2bf(float x) {
  unsigned int u = __float_as_uint(x);
  u += 0x7FFFu + ((u >> 16) & 1u);              // round-to-nearest-even
  return (unsigned short)(u >> 16);
}

__device__ __forceinline__ v8f wmma_bf16(v16bf a, v16bf b, v8f c) {
  // (neg_a, A, neg_b, B, c_mod, C, reuse_a, reuse_b)
  return __builtin_amdgcn_wmma_f32_16x16x32_bf16(false, a, false, b,
                                                 (short)0, c, false, false);
}

// Generic LDS pointer -> 32-bit LDS byte offset (low 32 bits of aperture addr).
__device__ __forceinline__ unsigned lds_off(const void* p) {
  return (unsigned)(unsigned long long)p;
}

// Async global -> LDS copy, 16 bytes per lane; tracked by ASYNCcnt.
__device__ __forceinline__ void async_ld_b128(unsigned lds, const void* g) {
  asm volatile("global_load_async_to_lds_b128 %0, %1, off"
               :: "v"(lds), "v"(g) : "memory");
}

// ---------------------------------------------------------------------------
// Embed: x[b,s,:] = (s==0 ? cls : feat[b, s-1]) + pos[s]; pad rows zero.
// ---------------------------------------------------------------------------
__global__ __launch_bounds__(256) void k_embed(
    const float* __restrict__ feat, const float* __restrict__ cls,
    const float* __restrict__ pos, float* __restrict__ xf,
    unsigned short* __restrict__ xb)
{
  size_t idx = (size_t)blockIdx.x * 256 + threadIdx.x;
  int d = (int)(idx & (DM - 1));
  int s = (int)((idx >> 9) & (SP - 1));
  int b = (int)(idx >> 19);
  float v = 0.f;
  if (s < SEQ) {
    float base = (s == 0) ? cls[d] : feat[((size_t)b * LPTS + (s - 1)) * DM + d];
    v = base + pos[(size_t)s * DM + d];
  }
  xf[idx] = v;
  xb[idx] = f2bf(v);
}

// ---------------------------------------------------------------------------
// f32 -> bf16 convert (weights)
// ---------------------------------------------------------------------------
__global__ __launch_bounds__(256) void k_cvt(
    const float* __restrict__ in, unsigned short* __restrict__ out, int n)
{
  int i = blockIdx.x * 256 + threadIdx.x;
  if (i < n) out[i] = f2bf(in[i]);
}

// ---------------------------------------------------------------------------
// WMMA GEMM: C[m,n] = act( sum_k A[m,k] * W[n,k] + bias[n] )
// A bf16 [M,K] lda; W bf16 [N,K] ldb (PyTorch linear layout).
// 64x64 block tile, 8 waves x two 16x16 accumulators.
// Double-buffered async LDS staging (global_load_async_to_lds_b128).
// Grid must tile M,N exactly (multiples of 64); K multiple of 32.
// ---------------------------------------------------------------------------
__global__ __launch_bounds__(256) void k_gemm_nt(
    const unsigned short* __restrict__ A, int lda,
    const unsigned short* __restrict__ W, int ldb,
    const float* __restrict__ bias,
    float* __restrict__ Cf, unsigned short* __restrict__ Cb, int ldc,
    int K, int relu)
{
  // 16 uints used + 4 pad: 80B rows => 16B aligned for b128, conflict-free.
  __shared__ __align__(16) unsigned int As[2][64][20];
  __shared__ __align__(16) unsigned int Bs[2][64][20];

  int tid  = threadIdx.x;
  int wave = tid >> 5, lane = tid & 31;
  int hi   = lane >> 4, l16 = lane & 15;
  int row0 = blockIdx.y * 64, col0 = blockIdx.x * 64;
  int n_t  = wave & 3;
  int m_t0 = (wave >> 2) << 1;          // wave covers m-tiles m_t0, m_t0+1

  // Staging: thread -> (row, 16B segment); one b128 per tile per array.
  int rS = tid >> 2, sS = tid & 3;      // 64 rows x 4 segs (8 bf16 each)
  const unsigned short* pA = A + (size_t)(row0 + rS) * lda + sS * 8;
  const unsigned short* pB = W + (size_t)(col0 + rS) * ldb + sS * 8;
  unsigned ldsA[2] = { lds_off(&As[0][rS][sS * 4]), lds_off(&As[1][rS][sS * 4]) };
  unsigned ldsB[2] = { lds_off(&Bs[0][rS][sS * 4]), lds_off(&Bs[1][rS][sS * 4]) };

  AccF acc0, acc1;
#pragma unroll
  for (int r = 0; r < 8; ++r) { acc0.f[r] = 0.f; acc1.f[r] = 0.f; }

  // Prologue: stage tile 0 into buffer 0.
  async_ld_b128(ldsA[0], pA);
  async_ld_b128(ldsB[0], pB);

  int ntile = K >> 5;
  int buf = 0;
  for (int i = 0; i < ntile; ++i) {
    pA += 32; pB += 32;
    if (i + 1 < ntile) {
      __builtin_prefetch(pA + 32, 0, 1);     // L2 warm, 2 tiles ahead (speculative)
      async_ld_b128(ldsA[buf ^ 1], pA);      // stage tile i+1 into other buffer
      async_ld_b128(ldsB[buf ^ 1], pB);
      asm volatile("s_wait_asynccnt 0x2" ::: "memory");  // tile i landed
    } else {
      asm volatile("s_wait_asynccnt 0x0" ::: "memory");
    }
    __syncthreads();

    FragBF a0, a1, bfr;
    int mr0 = (m_t0 << 4) + l16;
    int nr  = (n_t  << 4) + l16;
#pragma unroll
    for (int v = 0; v < 8; ++v) {
      // A-fragment pair index per ISA 16-bit 16x32 layout
      int pi = (v & 3) + ((v >> 2) << 3) + (hi << 2);
      a0.u[v] = As[buf][mr0][pi];
      a1.u[v] = As[buf][mr0 + 16][pi];
      // B-fragment (32x16): same K pair index as A fragment
      bfr.u[v] = Bs[buf][nr][pi];
    }
    acc0.v = wmma_bf16(a0.h, bfr.h, acc0.v);
    acc1.v = wmma_bf16(a1.h, bfr.h, acc1.v);
    __syncthreads();                          // done reading buf before overwrite
    buf ^= 1;
  }

  int n = col0 + (n_t << 4) + l16;
  float bs = bias ? bias[n] : 0.f;
#pragma unroll
  for (int r = 0; r < 8; ++r) {
    int m0 = row0 + (m_t0 << 4) + (hi << 3) + r;   // C layout: lane n, row hi*8+r
    int m1 = m0 + 16;
    float c0 = acc0.f[r] + bs;
    float c1 = acc1.f[r] + bs;
    if (relu) { c0 = fmaxf(c0, 0.f); c1 = fmaxf(c1, 0.f); }
    if (Cf) { Cf[(size_t)m0 * ldc + n] = c0; Cf[(size_t)m1 * ldc + n] = c1; }
    if (Cb) { Cb[(size_t)m0 * ldc + n] = f2bf(c0); Cb[(size_t)m1 * ldc + n] = f2bf(c1); }
  }
}

// ---------------------------------------------------------------------------
// Flash attention. qkv bf16 [B*Sp, 3D]. Grid: (Sp/128, H, B). 8 waves, each
// wave owns 16 query rows. K/V tiles staged by async b128 copies; online
// softmax over 32-key blocks; scores never touch memory.
// ---------------------------------------------------------------------------
__global__ __launch_bounds__(256) void k_flash(
    const unsigned short* __restrict__ qkv, unsigned short* __restrict__ O)
{
  __shared__ __align__(16) unsigned int Kl[32][36];  // 32 keys x 64 bf16 (+pad)
  __shared__ __align__(16) unsigned int Vl[32][36];  // 32 keys x 64 bf16 (+pad)
  __shared__ unsigned short Ps[8][16][32];           // per-wave P tile 16x32 bf16

  const unsigned int* q32 = (const unsigned int*)qkv;
  int tid  = threadIdx.x;
  int wave = tid >> 5, lane = tid & 31;
  int hi   = lane >> 4, l16 = lane & 15;
  int head = blockIdx.y, batch = blockIdx.z;
  int qrow0 = blockIdx.x * 128 + wave * 16;          // within Sp

  // Q fragments (two 16x32 k-chunks of the 16x64 Q tile), direct from global.
  size_t baseQ = (((size_t)(batch * SP + qrow0 + l16) * (3 * DM)) + head * HD) >> 1;
  FragBF q0, q1;
#pragma unroll
  for (int v = 0; v < 8; ++v) {
    int pi = (v & 3) + ((v >> 2) << 3) + (hi << 2);
    q0.u[v] = q32[baseQ + pi];
    q1.u[v] = q32[baseQ + 16 + pi];
  }

  // Async staging: thread -> one K b128 and one V b128 per key block.
  int rS = tid >> 3, sS = tid & 7;                   // 32 rows x 8 segs
  const unsigned short* pK =
      qkv + (size_t)(batch * SP + rS) * (3 * DM) + DM + head * HD + sS * 8;
  const unsigned short* pV = pK + DM;
  unsigned ldsK = lds_off(&Kl[rS][sS * 4]);
  unsigned ldsV = lds_off(&Vl[rS][sS * 4]);

  AccF o[4];
  float mrow[8], lrow[8];
#pragma unroll
  for (int j = 0; j < 4; ++j)
#pragma unroll
    for (int r = 0; r < 8; ++r) o[j].f[r] = 0.f;
#pragma unroll
  for (int r = 0; r < 8; ++r) { mrow[r] = -3.0e38f; lrow[r] = 0.f; }

  for (int kb = 0; kb < SEQ; kb += 32) {
    async_ld_b128(ldsK, pK);
    async_ld_b128(ldsV, pV);
    asm volatile("s_wait_asynccnt 0x0" ::: "memory");
    __syncthreads();

    // Scores: two 16x16 tiles covering 32 keys, K-dim 64 = 2 WMMAs each.
    AccF s[2];
#pragma unroll
    for (int nt = 0; nt < 2; ++nt) {
#pragma unroll
      for (int r = 0; r < 8; ++r) s[nt].f[r] = 0.f;
      FragBF kf;
      int nr = (nt << 4) + l16;
#pragma unroll
      for (int v = 0; v < 8; ++v) kf.u[v] = Kl[nr][(v & 3) + ((v >> 2) << 3) + (hi << 2)];
      s[nt].v = wmma_bf16(q0.h, kf.h, s[nt].v);
#pragma unroll
      for (int v = 0; v < 8; ++v) kf.u[v] = Kl[nr][16 + (v & 3) + ((v >> 2) << 3) + (hi << 2)];
      s[nt].v = wmma_bf16(q1.h, kf.h, s[nt].v);
      int kidx = kb + (nt << 4) + l16;
      bool valid = (kidx < SEQ);
#pragma unroll
      for (int r = 0; r < 8; ++r)
        s[nt].f[r] = valid ? s[nt].f[r] * 0.125f : -3.0e38f;  // scale = 1/sqrt(64)
    }

    // Online softmax per row (row r lives on the 16 lanes sharing hi).
#pragma unroll
    for (int r = 0; r < 8; ++r) {
      float mx = fmaxf(s[0].f[r], s[1].f[r]);
      mx = fmaxf(mx, __shfl_xor(mx, 1));
      mx = fmaxf(mx, __shfl_xor(mx, 2));
      mx = fmaxf(mx, __shfl_xor(mx, 4));
      mx = fmaxf(mx, __shfl_xor(mx, 8));
      float mn = fmaxf(mrow[r], mx);
      float al = __expf(mrow[r] - mn);
      float p0 = __expf(s[0].f[r] - mn);
      float p1 = __expf(s[1].f[r] - mn);
      float sum = p0 + p1;
      sum += __shfl_xor(sum, 1);
      sum += __shfl_xor(sum, 2);
      sum += __shfl_xor(sum, 4);
      sum += __shfl_xor(sum, 8);
      lrow[r] = lrow[r] * al + sum;
      mrow[r] = mn;
#pragma unroll
      for (int j = 0; j < 4; ++j) o[j].f[r] *= al;
      // Spill P tile (C-layout) to LDS for re-layout into an A-fragment.
      Ps[wave][(hi << 3) + r][l16]      = f2bf(p0);
      Ps[wave][(hi << 3) + r][16 + l16] = f2bf(p1);
    }
    asm volatile("s_wait_dscnt 0x0" ::: "memory");  // DS write -> DS read, same wave

    // Gather P as A-fragment (16x32), then O += P @ V for 4 column tiles.
    FragBF pf;
    const unsigned int* Pu = (const unsigned int*)(&Ps[wave][0][0]);
#pragma unroll
    for (int v = 0; v < 8; ++v) {
      int pi = (v & 3) + ((v >> 2) << 3) + (hi << 2);
      pf.u[v] = Pu[l16 * 16 + pi];
    }
#pragma unroll
    for (int j = 0; j < 4; ++j) {
      FragBF vf;
      int c2  = (j << 3) + (l16 >> 1);             // (col)>>1 in uint units
      bool odd = (l16 & 1) != 0;
#pragma unroll
      for (int v = 0; v < 8; ++v) {
        int k = ((v & 3) << 1) + ((v >> 2) << 4) + (hi << 3);  // B layout: K = 2(v&3) + 16(v>>2) + 8hi
        unsigned aa = Vl[k][c2];
        unsigned bb = Vl[k + 1][c2];
        vf.u[v] = odd ? ((aa >> 16) | (bb & 0xffff0000u))
                      : ((aa & 0xffffu) | (bb << 16));
      }
      o[j].v = wmma_bf16(pf.h, vf.h, o[j].v);
    }
    __syncthreads();                                // all done before next stage
    pK += (size_t)32 * (3 * DM);
    pV += (size_t)32 * (3 * DM);
  }

  // Normalize and write O (bf16) at head column offset.
#pragma unroll
  for (int j = 0; j < 4; ++j) {
    int col = head * HD + (j << 4) + l16;
#pragma unroll
    for (int r = 0; r < 8; ++r) {
      int m = batch * SP + qrow0 + (hi << 3) + r;
      O[(size_t)m * DM + col] = f2bf(o[j].f[r] / lrow[r]);
    }
  }
}

// ---------------------------------------------------------------------------
// Fused residual add + LayerNorm. One block per row. Pad rows -> zeros.
// ---------------------------------------------------------------------------
__global__ __launch_bounds__(256) void k_add_ln(
    const float* __restrict__ x, const float* __restrict__ t,
    const float* __restrict__ g, const float* __restrict__ bta,
    float* __restrict__ yf, unsigned short* __restrict__ yb)
{
  __shared__ float rs[256], rq[256];
  int row = blockIdx.x;
  int s = row & (SP - 1);
  int tid = threadIdx.x;
  size_t off = (size_t)row * DM;
  if (s >= SEQ) {
    yf[off + tid] = 0.f; yf[off + tid + 256] = 0.f;
    yb[off + tid] = 0;   yb[off + tid + 256] = 0;
    return;
  }
  float a = x[off + tid] + t[off + tid];
  float b = x[off + tid + 256] + t[off + tid + 256];
  rs[tid] = a + b;
  rq[tid] = a * a + b * b;
  __syncthreads();
  for (int st = 128; st > 0; st >>= 1) {
    if (tid < st) { rs[tid] += rs[tid + st]; rq[tid] += rq[tid + st]; }
    __syncthreads();
  }
  float mean = rs[0] * (1.f / DM);
  float var  = rq[0] * (1.f / DM) - mean * mean;
  float rstd = rsqrtf(var + 1e-5f);
  float y0 = (a - mean) * rstd * g[tid] + bta[tid];
  float y1 = (b - mean) * rstd * g[tid + 256] + bta[tid + 256];
  yf[off + tid] = y0;       yf[off + tid + 256] = y1;
  yb[off + tid] = f2bf(y0); yb[off + tid + 256] = f2bf(y1);
}

// ---------------------------------------------------------------------------
// Final LN of the CLS row of each sequence -> d_out [B, D] f32.
// ---------------------------------------------------------------------------
__global__ __launch_bounds__(256) void k_final(
    const float* __restrict__ x, const float* __restrict__ g,
    const float* __restrict__ bta, float* __restrict__ out)
{
  __shared__ float rs[256], rq[256];
  int b = blockIdx.x;
  int tid = threadIdx.x;
  size_t off = (size_t)b * SP * DM;       // CLS is row s=0
  float a0 = x[off + tid], a1 = x[off + tid + 256];
  rs[tid] = a0 + a1;
  rq[tid] = a0 * a0 + a1 * a1;
  __syncthreads();
  for (int st = 128; st > 0; st >>= 1) {
    if (tid < st) { rs[tid] += rs[tid + st]; rq[tid] += rq[tid + st]; }
    __syncthreads();
  }
  float mean = rs[0] * (1.f / DM);
  float var  = rq[0] * (1.f / DM) - mean * mean;
  float rstd = rsqrtf(var + 1e-5f);
  out[(size_t)b * DM + tid]       = (a0 - mean) * rstd * g[tid] + bta[tid];
  out[(size_t)b * DM + tid + 256] = (a1 - mean) * rstd * g[tid + 256] + bta[tid + 256];
}

// ---------------------------------------------------------------------------
// Host orchestration.
// Input order: 0 features, 1 batch_indices, 2 cls_token, 3 pos_emb,
//   per layer (x2): in_w,in_b,out_w,out_b,w1,b1,w2,b2,ln1_g,ln1_b,ln2_g,ln2_b,
//   then final_g, final_b.
// ---------------------------------------------------------------------------
extern "C" void kernel_launch(void* const* d_in, const int* in_sizes, int n_in,
                              void* d_out, int out_size, void* d_ws, size_t ws_size,
                              hipStream_t stream)
{
  (void)in_sizes; (void)n_in; (void)out_size; (void)ws_size;
  const float* feat = (const float*)d_in[0];
  const float* cls  = (const float*)d_in[2];
  const float* pos  = (const float*)d_in[3];
  const float* fing = (const float*)d_in[4 + 2 * 12 + 0];
  const float* finb = (const float*)d_in[4 + 2 * 12 + 1];

  // Workspace layout (~292 MB): FFN hidden reuses the QKV region.
  char* p = (char*)d_ws;
  float*          x_f32  = (float*)p;          p += (size_t)MROWS * DM * 4;
  float*          t_f32  = (float*)p;          p += (size_t)MROWS * DM * 4;
  unsigned short* x_bf   = (unsigned short*)p; p += (size_t)MROWS * DM * 2;
  unsigned short* o_bf   = (unsigned short*)p; p += (size_t)MROWS * DM * 2;
  unsigned short* qkv_bf = (unsigned short*)p; p += (size_t)MROWS * 3 * DM * 2;
  unsigned short* h_bf   = qkv_bf;             // reuse (disjoint lifetime)
  unsigned short* wbf    = (unsigned short*)p; // per-layer weight bf16 (reused)

  k_embed<<<dim3((MROWS * DM) / 256), 256, 0, stream>>>(feat, cls, pos, x_f32, x_bf);

  for (int l = 0; l < 2; ++l) {
    const float* in_w  = (const float*)d_in[4 + l * 12 + 0];
    const float* in_b  = (const float*)d_in[4 + l * 12 + 1];
    const float* out_w = (const float*)d_in[4 + l * 12 + 2];
    const float* out_b = (const float*)d_in[4 + l * 12 + 3];
    const float* w1    = (const float*)d_in[4 + l * 12 + 4];
    const float* b1    = (const float*)d_in[4 + l * 12 + 5];
    const float* w2    = (const float*)d_in[4 + l * 12 + 6];
    const float* b2    = (const float*)d_in[4 + l * 12 + 7];
    const float* g1    = (const float*)d_in[4 + l * 12 + 8];
    const float* be1   = (const float*)d_in[4 + l * 12 + 9];
    const float* g2    = (const float*)d_in[4 + l * 12 + 10];
    const float* be2   = (const float*)d_in[4 + l * 12 + 11];

    unsigned short* inw_bf  = wbf;
    unsigned short* outw_bf = inw_bf  + 3 * DM * DM;
    unsigned short* w1_bf   = outw_bf + DM * DM;
    unsigned short* w2_bf   = w1_bf   + DFF2 * DM;

    k_cvt<<<dim3((3 * DM * DM) / 256), 256, 0, stream>>>(in_w, inw_bf, 3 * DM * DM);
    k_cvt<<<dim3((DM * DM) / 256), 256, 0, stream>>>(out_w, outw_bf, DM * DM);
    k_cvt<<<dim3((DFF2 * DM) / 256), 256, 0, stream>>>(w1, w1_bf, DFF2 * DM);
    k_cvt<<<dim3((DM * DFF2) / 256), 256, 0, stream>>>(w2, w2_bf, DM * DFF2);

    // QKV: [M,512] x [1536,512]^T -> bf16 [M,1536]
    k_gemm_nt<<<dim3((3 * DM) / 64, MROWS / 64), 256, 0, stream>>>(
        x_bf, DM, inw_bf, DM, in_b, nullptr, qkv_bf, 3 * DM, DM, 0);

    // Flash attention -> O bf16 [M,512]
    k_flash<<<dim3(SP / 128, NH, BATCH), 256, 0, stream>>>(qkv_bf, o_bf);

    // Output projection -> t f32
    k_gemm_nt<<<dim3(DM / 64, MROWS / 64), 256, 0, stream>>>(
        o_bf, DM, outw_bf, DM, out_b, t_f32, nullptr, DM, DM, 0);

    k_add_ln<<<dim3(MROWS), 256, 0, stream>>>(x_f32, t_f32, g1, be1, x_f32, x_bf);

    // FFN1 (ReLU) -> bf16 hidden [M,1024]
    k_gemm_nt<<<dim3(DFF2 / 64, MROWS / 64), 256, 0, stream>>>(
        x_bf, DM, w1_bf, DM, b1, nullptr, h_bf, DFF2, DM, 1);

    // FFN2 -> t f32
    k_gemm_nt<<<dim3(DM / 64, MROWS / 64), 256, 0, stream>>>(
        h_bf, DFF2, w2_bf, DFF2, b2, t_f32, nullptr, DM, DFF2, 0);

    k_add_ln<<<dim3(MROWS), 256, 0, stream>>>(x_f32, t_f32, g2, be2, x_f32, x_bf);
  }

  k_final<<<dim3(BATCH), 256, 0, stream>>>(x_f32, fing, finb, (float*)d_out);
}
